// TemporalConvNet_39144331936370
// MI455X (gfx1250) — matrix operand order repaired and checked by reading, not compiled, not run on hardware
//
#include <hip/hip_runtime.h>

#define TT 128          // time steps
#define CC 128          // channels
#define FF 76           // input features
#define PITCH 136u      // f16 elements per LDS row (128 + 8 pad -> conflict-free)
#define BUFE (TT * PITCH)

typedef _Float16 v16h __attribute__((ext_vector_type(16)));
typedef _Float16 v8h  __attribute__((ext_vector_type(8)));
typedef float    v8f  __attribute__((ext_vector_type(8)));

// Explicit LDS (addrspace 3) types: 32-bit ds addressing, no generic-ptr null checks.
typedef __attribute__((address_space(3))) _Float16 lds_h;
typedef __attribute__((address_space(3))) v8h      lds_v8h;

// ---- A-matrix fragment (16x32 f16, weights from global fp32) ----
// Layout (ISA 7.12.2): lane = M (mod 16); elems 0..7 -> K = kbase+(hi?8:0)+i,
// elems 8..15 -> K = kbase+16+(hi?8:0)+i.
__device__ __forceinline__ v16h load_a_frag(const float* __restrict__ W,
                                            int Cin, int kstride, int tap,
                                            int co, int kbase, int hiA) {
  const float* row = W + (size_t)co * (size_t)Cin * (size_t)kstride + tap;
  v16h a;
#pragma unroll
  for (int i = 0; i < 8; ++i) {
    int k0 = kbase + hiA + i;
    int k1 = kbase + 16 + hiA + i;
    a[i]     = (k0 < Cin) ? (_Float16)row[(size_t)k0 * kstride] : (_Float16)0.0f;
    a[i + 8] = (k1 < Cin) ? (_Float16)row[(size_t)k1 * kstride] : (_Float16)0.0f;
  }
  return a;
}

// ---- B-matrix fragment (32x16 f16 from LDS, time-major [t][ci]) ----
// lane = N (t, mod 16); elems 0..7 -> K = kbase + (hi?8:0) + i, elems 8..15 -> K = kbase + 16 + (hi?8:0) + i.
__device__ __forceinline__ v16h load_b_frag(const lds_h* buf, int kbase, int t,
                                            int hiB) {
  const lds_h* p = buf + (unsigned)t * PITCH + (unsigned)(kbase + (hiB >> 1));
  v8h u0 = *(const lds_v8h*)(p);       // ds_load_b128, 16B aligned
  v8h u1 = *(const lds_v8h*)(p + 16);
  v16h b;
#pragma unroll
  for (int i = 0; i < 8; ++i) { b[i] = u0[i]; b[i + 8] = u1[i]; }
  return b;
}

// Branch-free guarded variant for the (rare) tiles crossing t=0: clamp + select.
__device__ __forceinline__ v16h load_b_guard(const lds_h* buf, int kbase, int t,
                                             int hiB) {
  int tc = t < 0 ? 0 : t;
  v16h b = load_b_frag(buf, kbase, tc, hiB);
  if (t < 0) {
#pragma unroll
    for (int i = 0; i < 16; ++i) b[i] = (_Float16)0.0f;   // -> v_cndmask x8
  }
  return b;
}

// One causal conv (K=2, dilation dil): dst = relu( relu(conv(src)+bias) + res )
// HAS_RES : identity residual from resbuf (may alias dst -> in-place)
// HAS_DOWN: residual = Wd @ x0 + bd (level-0 downsample GEMM)
template <bool HAS_RES, bool HAS_DOWN>
__device__ void conv_block(const lds_h* src, lds_h* dst, const lds_h* resbuf,
                           const float* __restrict__ W, const float* __restrict__ bias,
                           int Cin, int dil,
                           const float* __restrict__ Wd, const float* __restrict__ bd,
                           const lds_h* x0, int CinD) {
  __syncthreads();
  const int tid  = threadIdx.x;
  const int lane = tid & 31;
  const int wave = tid >> 5;
  const int lo   = lane & 15;
  const int hi   = lane >> 4;
  const int hiA8  = hi ? 8 : 0;
  const int hiB16 = hi ? 16 : 0;
  const int hiM8  = hi ? 8 : 0;
  const int m0 = wave * 16;          // 8 waves cover Cout = 128

  // Preload weight fragments once per wave, reuse across all 8 N-tiles.
  v16h a0[4], a1[4], ad[4];
#pragma unroll
  for (int kt = 0; kt < 4; ++kt) {
    a0[kt] = load_a_frag(W, Cin, 2, 0, m0 + lo, kt * 32, hiA8);  // past tap
    a1[kt] = load_a_frag(W, Cin, 2, 1, m0 + lo, kt * 32, hiA8);  // current tap
    if (HAS_DOWN) ad[kt] = load_a_frag(Wd, CinD, 1, 0, m0 + lo, kt * 32, hiA8);
  }
  float bb[8], bdv[8];
#pragma unroll
  for (int r = 0; r < 8; ++r) {
    bb[r]  = bias[m0 + hiM8 + r];
    bdv[r] = HAS_DOWN ? bd[m0 + hiM8 + r] : 0.0f;
  }

  for (int n0 = 0; n0 < TT; n0 += 16) {
    const int t = n0 + lo;
    v8f acc;
#pragma unroll
    for (int r = 0; r < 8; ++r) acc[r] = 0.0f;

    if (n0 >= dil) {
      // Uniform-safe path, software-pipelined: issue K-step kt+1's fragment
      // loads before the WMMAs of step kt so LDS latency hides under the
      // matrix pipe (partial s_wait_dscnt instead of full waits).
      v16h bc = load_b_frag(src, 0, t, hiB16);
      v16h bp = load_b_frag(src, 0, t - dil, hiB16);
#pragma unroll
      for (int kt = 0; kt < 4; ++kt) {
        v16h bc2, bp2;
        if (kt < 3) {
          bc2 = load_b_frag(src, (kt + 1) * 32, t, hiB16);
          bp2 = load_b_frag(src, (kt + 1) * 32, t - dil, hiB16);
        }
        acc = __builtin_amdgcn_wmma_f32_16x16x32_f16(false, a1[kt], false, bc,
                                                     (short)0, acc, false, false);
        acc = __builtin_amdgcn_wmma_f32_16x16x32_f16(false, a0[kt], false, bp,
                                                     (short)0, acc, false, false);
        bc = bc2;
        bp = bp2;
      }
    } else {                    // boundary tiles: branch-free clamped loads
#pragma unroll
      for (int kt = 0; kt < 4; ++kt) {
        v16h bcur  = load_b_frag(src, kt * 32, t, hiB16);
        v16h bpast = load_b_guard(src, kt * 32, t - dil, hiB16);
        acc = __builtin_amdgcn_wmma_f32_16x16x32_f16(false, a1[kt], false, bcur,
                                                     (short)0, acc, false, false);
        acc = __builtin_amdgcn_wmma_f32_16x16x32_f16(false, a0[kt], false, bpast,
                                                     (short)0, acc, false, false);
      }
    }

    v8f accD;
    if (HAS_DOWN) {
#pragma unroll
      for (int r = 0; r < 8; ++r) accD[r] = 0.0f;
      v16h bx = load_b_frag(x0, 0, t, hiB16);
#pragma unroll
      for (int kt = 0; kt < 4; ++kt) {
        v16h bx2;
        if (kt < 3) bx2 = load_b_frag(x0, (kt + 1) * 32, t, hiB16);
        accD = __builtin_amdgcn_wmma_f32_16x16x32_f16(false, ad[kt], false, bx,
                                                      (short)0, accD, false, false);
        bx = bx2;
      }
    }

    // Epilogue: D tile (M=co, N=t); lane owns row t, cols m0+hiM8+0..7.
    const unsigned off = (unsigned)t * PITCH + (unsigned)(m0 + hiM8);
    v8h res8;
    if (HAS_RES) res8 = *(const lds_v8h*)(resbuf + off);
    v8h s;
#pragma unroll
    for (int r = 0; r < 8; ++r) {
      float h  = fmaxf(acc[r] + bb[r], 0.0f);   // relu(conv + b)
      float rv = 0.0f;
      if (HAS_RES)  rv = (float)res8[r];
      if (HAS_DOWN) rv = accD[r] + bdv[r];
      s[r] = (_Float16)fmaxf(h + rv, 0.0f);
    }
    *(lds_v8h*)(dst + off) = s;
  }
}

__global__ __launch_bounds__(256)
void tcn_fused_kernel(const float* __restrict__ x,
                      const float* __restrict__ W1_0, const float* __restrict__ b1_0,
                      const float* __restrict__ W2_0, const float* __restrict__ b2_0,
                      const float* __restrict__ Wd,  const float* __restrict__ bd,
                      const float* __restrict__ W1s, const float* __restrict__ b1s,
                      const float* __restrict__ W2s, const float* __restrict__ b2s,
                      float* __restrict__ out) {
  __shared__ _Float16 smem_[3 * BUFE];   // ~102 KB of WGP LDS
  lds_h* smem = (lds_h*)smem_;
  lds_h* buf0 = smem;
  lds_h* buf1 = smem + BUFE;
  lds_h* buf2 = smem + 2 * BUFE;

  const int b   = blockIdx.x;
  const int tid = threadIdx.x;

  // Stage x[b] time-major into LDS, zero-pad channels 76..127 (coalesced reads).
  const float* xb = x + (size_t)b * TT * FF;
  for (int idx = tid; idx < TT * CC; idx += 256) {
    int t = idx >> 7, ci = idx & 127;
    buf0[(unsigned)t * PITCH + (unsigned)ci] =
        (ci < FF) ? (_Float16)xb[t * FF + ci] : (_Float16)0.0f;
  }
  // Warm L2/WGP$ with level-1 weights while level 0 runs.
  for (int off = tid * 64; off < CC * CC * 2; off += 256 * 64) {
    __builtin_prefetch(W1s + off, 0, 1);
    __builtin_prefetch(W2s + off, 0, 1);
  }

  // Level 0: F->C with downsample-GEMM residual.
  conv_block<false, false>(buf0, buf1, buf0, W1_0, b1_0, FF, 1,
                           nullptr, nullptr, buf0, 0);
  conv_block<false, true >(buf1, buf2, buf0, W2_0, b2_0, CC, 1,
                           Wd, bd, buf0, FF);

  // Levels 1..5: identity residual, conv2 writes X in place.
  lds_h* X = buf2;
  lds_h* H = buf0;
  for (int lvl = 0; lvl < 5; ++lvl) {
    const int dil = 2 << lvl;                       // 2,4,8,16,32
    const float* W1 = W1s + (size_t)lvl * CC * CC * 2;
    const float* B1 = b1s + (size_t)lvl * CC;
    const float* W2 = W2s + (size_t)lvl * CC * CC * 2;
    const float* B2 = b2s + (size_t)lvl * CC;
    if (lvl < 4) {  // prefetch next level's weights into cache
      for (int off = tid * 64; off < CC * CC * 2; off += 256 * 64) {
        __builtin_prefetch(W1 + (size_t)CC * CC * 2 + off, 0, 1);
        __builtin_prefetch(W2 + (size_t)CC * CC * 2 + off, 0, 1);
      }
    }
    conv_block<false, false>(X, H, X, W1, B1, CC, dil, nullptr, nullptr, X, 0);
    conv_block<true,  false>(H, X, X, W2, B2, CC, dil, nullptr, nullptr, X, 0);
  }

  // Cumulative mean over time, write [B,T,C] fp32 (coalesced over c).
  __syncthreads();
  if (tid < CC) {
    const int c = tid;
    float s = 0.0f;
    float* ob = out + (size_t)b * TT * CC;
    for (int t = 0; t < TT; ++t) {
      s += (float)X[(unsigned)t * PITCH + (unsigned)c];
      ob[t * CC + c] = s / (float)(t + 1);
    }
  }
}

extern "C" void kernel_launch(void* const* d_in, const int* in_sizes, int n_in,
                              void* d_out, int out_size, void* d_ws, size_t ws_size,
                              hipStream_t stream) {
  const float* x    = (const float*)d_in[0];
  const float* W1_0 = (const float*)d_in[1];
  const float* b1_0 = (const float*)d_in[2];
  const float* W2_0 = (const float*)d_in[3];
  const float* b2_0 = (const float*)d_in[4];
  const float* Wd   = (const float*)d_in[5];
  const float* bd   = (const float*)d_in[6];
  const float* W1s  = (const float*)d_in[7];
  const float* b1s  = (const float*)d_in[8];
  const float* W2s  = (const float*)d_in[9];
  const float* b2s  = (const float*)d_in[10];
  float* out = (float*)d_out;

  tcn_fused_kernel<<<dim3(128), dim3(256), 0, stream>>>(
      x, W1_0, b1_0, W2_0, b2_0, Wd, bd, W1s, b1s, W2s, b2s, out);
}
